// MILPGNNModel_31748398252366
// MI455X (gfx1250) — matrix operand order repaired and checked by reading, not compiled, not run on hardware
//
#include <hip/hip_runtime.h>
#include <hip/hip_bf16.h>
#include <math.h>

#define V_N 50000
#define C_N 20000
#define E_N 500000
#define H_N 256

typedef __attribute__((ext_vector_type(16))) __bf16 v16bf;
typedef __attribute__((ext_vector_type(2)))  __bf16 v2bf;
typedef __attribute__((ext_vector_type(8)))  float  v8f;

// ---------------------------------------------------------------------------
// Pack W (K x 256 f32 row-major) into bf16 WMMA B-fragment order.
// Layout: pb[kb][ntile][lane][e], value = W[kb*32 + (lane/16)*16 + e][ntile*16 + lane%16]
// (16-bit B 32x16, contiguous k order used here (same as A): lanes 0-15 hold K=0-15, lanes 16-31 hold K=16-31; N = lane%16)
// ---------------------------------------------------------------------------
__global__ void pack_b_kernel(const float* __restrict__ W, __bf16* __restrict__ pb, int K) {
    int p = blockIdx.x * 256 + threadIdx.x;
    if (p >= K * 256) return;
    int e    = p & 15;
    int lane = (p >> 4) & 31;
    int nt   = (p >> 9) & 15;
    int kb   = p >> 13;
    int k = kb * 32 + (lane >> 4) * 16 + e;
    int n = nt * 16 + (lane & 15);
    pb[p] = (__bf16)W[(size_t)k * H_N + n];
}

// ---------------------------------------------------------------------------
// Y = epilogue(A0 @ W[0:K0] + A1 @ W[K0:K0+K1] + bias),  W packed as pb (bf16)
// epilogue: optional LayerNorm(g,bt)+ReLU; optional residual out = resin + Y.
// Block: 256 threads = 8 waves; tile = 64 rows x 256 cols.
// wave w: row-tile = w%4, col-half = w/4 (8 col-tiles of 16).
// Double-buffered LDS A staging (float2-wide) + preloaded B fragments.
// ---------------------------------------------------------------------------
__launch_bounds__(256, 1)
__global__ void gemm_wmma_kernel(const float* __restrict__ A0, int K0,
                                 const float* __restrict__ A1, int K1,
                                 const __bf16* __restrict__ pb,
                                 const float* __restrict__ bias,
                                 const float* __restrict__ g,
                                 const float* __restrict__ bt,
                                 const float* __restrict__ resin,
                                 float* __restrict__ resout,
                                 float* __restrict__ out,
                                 int nrows, int doLN) {
    __shared__ __bf16 sA[2][64][40];       // double-buffered 64x32 bf16 tile (padded)
    __shared__ float  ssum[64][2];
    __shared__ float  ssq[64][2];

    const int t     = threadIdx.x;
    const int wave  = t >> 5;
    const int lane  = t & 31;
    const int half  = lane >> 4;           // 0/1
    const int ln16  = lane & 15;
    const int rtile = wave & 3;
    const int chalf = wave >> 2;           // 0/1 -> cols [0,128) or [128,256)
    const int rowbase = blockIdx.x * 64;

    v8f acc[8];
    #pragma unroll
    for (int i = 0; i < 8; i++)
        #pragma unroll
        for (int q = 0; q < 8; q++) acc[i][q] = 0.f;

    const int kbt = (K0 + K1) >> 5;

    // staging geometry: lanes 0-15 -> row r, cols 2l..2l+1 ; lanes 16-31 -> row r+1
    const int srow = wave * 8 + half;      // + j*2 below
    const int scol = ln16 * 2;

    // fetch global A data for K-step kb into registers (4 x float2, coalesced 256B/wave)
    auto fetch = [&](int kb, float2* pf) {
        const float* A; int stride, kof;
        if (kb * 32 < K0) { A = A0; stride = K0; kof = kb * 32; }
        else              { A = A1; stride = K1; kof = kb * 32 - K0; }
        #pragma unroll
        for (int j = 0; j < 4; j++) {
            int grow = rowbase + srow + j * 2;
            if (grow < nrows)
                pf[j] = *(const float2*)(A + (size_t)grow * stride + kof + scol);
            else
                pf[j] = make_float2(0.f, 0.f);
        }
    };
    auto stage = [&](int buf, const float2* pf) {
        #pragma unroll
        for (int j = 0; j < 4; j++) {
            v2bf pk;
            pk[0] = (__bf16)pf[j].x;
            pk[1] = (__bf16)pf[j].y;
            *(v2bf*)&sA[buf][srow + j * 2][scol] = pk;
        }
    };

    // prologue: stage K-step 0
    {
        float2 pf[4];
        fetch(0, pf);
        stage(0, pf);
    }
    __syncthreads();

    for (int kb = 0; kb < kbt; ++kb) {
        const int buf = kb & 1;

        // prefetch next A tile into registers (loads in flight under the WMMAs)
        float2 nx[4];
        if (kb + 1 < kbt) fetch(kb + 1, nx);

        // A fragment (16-bit A 16x32, contiguous k order matching packed B):
        // K = half*16 + e;  M = rtile*16 + lane%16
        v16bf afrag;
        const int m = rtile * 16 + ln16;
        #pragma unroll
        for (int e = 0; e < 16; e++) {
            int k = half * 16 + e;
            afrag[e] = sA[buf][m][k];
        }

        // preload all 8 B fragments (clauses of b128 loads), then WMMA chain
        v16bf bfr[8];
        #pragma unroll
        for (int nt = 0; nt < 8; nt++) {
            int ntile = chalf * 8 + nt;
            bfr[nt] = *(const v16bf*)(pb + (((size_t)kb * 16 + ntile) << 9) + lane * 16);
        }
        #pragma unroll
        for (int nt = 0; nt < 8; nt++) {
            acc[nt] = __builtin_amdgcn_wmma_f32_16x16x32_bf16(
                false, afrag, false, bfr[nt], (short)0, acc[nt], false, false);
        }

        if (kb + 1 < kbt) {
            stage(buf ^ 1, nx);
            __syncthreads();
        }
    }

    // ---- epilogue: bias, LN stats (cross-lane + cross-wave), write ----
    #pragma unroll
    for (int nt = 0; nt < 8; nt++) {
        int col = chalf * 128 + nt * 16 + ln16;
        float b = bias[col];
        #pragma unroll
        for (int i = 0; i < 8; i++) acc[nt][i] += b;
    }

    float mean[8], rstd[8];
    if (doLN) {
        #pragma unroll
        for (int i = 0; i < 8; i++) {
            float s = 0.f, sq = 0.f;
            #pragma unroll
            for (int nt = 0; nt < 8; nt++) { float v = acc[nt][i]; s += v; sq += v * v; }
            // reduce over the 16 lanes of this half (same row, different cols)
            #pragma unroll
            for (int msk = 1; msk < 16; msk <<= 1) {
                s  += __shfl_xor(s,  msk);
                sq += __shfl_xor(sq, msk);
            }
            if (ln16 == 0) {
                int mrow = rtile * 16 + half * 8 + i;
                ssum[mrow][chalf] = s;
                ssq[mrow][chalf]  = sq;
            }
        }
        __syncthreads();
        #pragma unroll
        for (int i = 0; i < 8; i++) {
            int mrow = rtile * 16 + half * 8 + i;
            float s  = ssum[mrow][0] + ssum[mrow][1];
            float sq = ssq[mrow][0]  + ssq[mrow][1];
            float mu = s * (1.f / 256.f);
            mean[i] = mu;
            rstd[i] = rsqrtf(sq * (1.f / 256.f) - mu * mu + 1e-5f);
        }
    }

    #pragma unroll
    for (int nt = 0; nt < 8; nt++) {
        int col = chalf * 128 + nt * 16 + ln16;
        float gv  = doLN ? g[col]  : 1.f;
        float btv = doLN ? bt[col] : 0.f;
        #pragma unroll
        for (int i = 0; i < 8; i++) {
            int grow = rowbase + rtile * 16 + half * 8 + i;
            if (grow < nrows) {
                float v = acc[nt][i];
                if (doLN) v = fmaxf((v - mean[i]) * rstd[i] * gv + btv, 0.f);
                size_t idx = (size_t)grow * H_N + col;
                out[idx] = v;
                if (resout) resout[idx] = resin[idx] + v;
            }
        }
    }
}

// ---------------------------------------------------------------------------
// Tiny-K first MLP layer: wave-per-row, out = relu(LN(x @ W1 + b1))
// ---------------------------------------------------------------------------
template <int KIN>
__global__ void emb1_kernel(const float* __restrict__ X, int n,
                            const float* __restrict__ W1, const float* __restrict__ b1,
                            const float* __restrict__ g,  const float* __restrict__ bt,
                            float* __restrict__ out) {
    int wave = threadIdx.x >> 5, lane = threadIdx.x & 31;
    int row = blockIdx.x * 8 + wave;
    if (row >= n) return;
    float x[KIN];
    #pragma unroll
    for (int k = 0; k < KIN; k++) x[k] = X[(size_t)row * KIN + k];
    float y[8];
    #pragma unroll
    for (int j = 0; j < 8; j++) {
        int col = lane + 32 * j;
        float a = b1[col];
        #pragma unroll
        for (int k = 0; k < KIN; k++) a = fmaf(x[k], W1[k * H_N + col], a);
        y[j] = a;
    }
    float s = 0.f, sq = 0.f;
    #pragma unroll
    for (int j = 0; j < 8; j++) { s += y[j]; sq += y[j] * y[j]; }
    #pragma unroll
    for (int msk = 1; msk < 32; msk <<= 1) { s += __shfl_xor(s, msk); sq += __shfl_xor(sq, msk); }
    float mu = s * (1.f / 256.f);
    float rs = rsqrtf(sq * (1.f / 256.f) - mu * mu + 1e-5f);
    #pragma unroll
    for (int j = 0; j < 8; j++) {
        int col = lane + 32 * j;
        out[(size_t)row * H_N + col] = fmaxf((y[j] - mu) * rs * g[col] + bt[col], 0.f);
    }
}

// ---------------------------------------------------------------------------
__global__ void count_kernel(const int* __restrict__ idx, int* __restrict__ cnt, int n) {
    int i = blockIdx.x * 256 + threadIdx.x;
    if (i < n) atomicAdd(&cnt[idx[i]], 1);
}

// wave-per-edge: agg[dst] += lin[src] * sigmoid(ev*gw + gb)
__global__ void scatter_kernel(const float* __restrict__ lin,
                               const int* __restrict__ src,
                               const int* __restrict__ dst,
                               const float* __restrict__ ev,
                               const float* __restrict__ gw,
                               const float* __restrict__ gb,
                               float* __restrict__ agg, int ne) {
    int wave = threadIdx.x >> 5, lane = threadIdx.x & 31;
    int e = blockIdx.x * 8 + wave;
    if (e >= ne) return;
    int s = src[e], d = dst[e];
    float evv = ev[e];
    #pragma unroll
    for (int j = 0; j < 8; j++) {
        int c = lane + 32 * j;
        float gate = 1.f / (1.f + __expf(-(evv * gw[c] + gb[c])));
        float m = lin[(size_t)s * H_N + c] * gate;
        unsafeAtomicAdd(&agg[(size_t)d * H_N + c], m);
    }
}

__global__ void norm_kernel(float* __restrict__ agg, const int* __restrict__ cnt, int n) {
    int i = blockIdx.x * 256 + threadIdx.x;
    if (i < n * H_N) {
        int r = i >> 8;
        int c = cnt[r];
        agg[i] /= (float)(c > 1 ? c : 1);
    }
}

// ---------------------------------------------------------------------------
extern "C" void kernel_launch(void* const* d_in, const int* in_sizes, int n_in,
                              void* d_out, int out_size, void* d_ws, size_t ws_size,
                              hipStream_t stream) {
    const float* var_feats = (const float*)d_in[0];   // V x 21
    const float* con_feats = (const float*)d_in[1];   // C x 6
    const int*   edge_idx  = (const int*)d_in[2];     // 2 x E
    const float* edge_val  = (const float*)d_in[3];   // E
    const int* ci = edge_idx;
    const int* vi = edge_idx + E_N;

    int p = 4;
    auto nf = [&]() { return (const float*)d_in[p++]; };
    // params flattened in dict-literal (insertion) order
    const float *vw1=nf(), *vb1=nf(), *vg1=nf(), *vbt1=nf(), *vw2=nf(), *vb2=nf(), *vg2=nf(), *vbt2=nf();
    const float *cw1=nf(), *cb1=nf(), *cg1=nf(), *cbt1=nf(), *cw2=nf(), *cb2=nf(), *cg2=nf(), *cbt2=nf();
    struct Layer {
        const float *lw_vc,*lb_vc,*gw_vc,*gb_vc,*uw_vc,*ub_vc,*ug_vc,*ut_vc;
        const float *lw_cv,*lb_cv,*gw_cv,*gb_cv,*uw_cv,*ub_cv,*ug_cv,*ut_cv;
    } L[2];
    for (int l = 0; l < 2; l++) {
        L[l].lw_vc=nf(); L[l].lb_vc=nf(); L[l].gw_vc=nf(); L[l].gb_vc=nf();
        L[l].uw_vc=nf(); L[l].ub_vc=nf(); L[l].ug_vc=nf(); L[l].ut_vc=nf();
        L[l].lw_cv=nf(); L[l].lb_cv=nf(); L[l].gw_cv=nf(); L[l].gb_cv=nf();
        L[l].uw_cv=nf(); L[l].ub_cv=nf(); L[l].ug_cv=nf(); L[l].ut_cv=nf();
    }

    // ---- workspace carving (256B aligned) ----
    char* w = (char*)d_ws;
    auto carve = [&](size_t bytes) -> void* {
        void* r = (void*)w; w += (bytes + 255) & ~(size_t)255; return r;
    };
    float* lin  = (float*)carve((size_t)V_N * H_N * 4);  // scratch: emb1 out / msg linear / vh_new
    float* vh   = (float*)carve((size_t)V_N * H_N * 4);
    float* vh2  = (float*)carve((size_t)V_N * H_N * 4);
    float* aggv = (float*)carve((size_t)V_N * H_N * 4);
    float* ch   = (float*)carve((size_t)C_N * H_N * 4);
    float* ch2  = (float*)carve((size_t)C_N * H_N * 4);
    float* chn  = (float*)carve((size_t)C_N * H_N * 4);
    float* aggc = (float*)carve((size_t)C_N * H_N * 4);
    int*   cntv = (int*)carve((size_t)V_N * 4);
    int*   cntc = (int*)carve((size_t)C_N * 4);
    __bf16* pw_lin = (__bf16*)carve((size_t)256 * H_N * 2);  // packed 256x256
    __bf16* pw_up  = (__bf16*)carve((size_t)512 * H_N * 2);  // packed 512x256

    dim3 blk(256);
    auto gemm = [&](const float* A0, int K0, const float* A1, int K1,
                    const float* W, __bf16* pw, const float* b,
                    const float* gg, const float* bb,
                    const float* rin, float* rout, float* o, int n, int doLN) {
        int K = K0 + K1;
        pack_b_kernel<<<dim3(K), blk, 0, stream>>>(W, pw, K);
        gemm_wmma_kernel<<<dim3((n + 63) / 64), blk, 0, stream>>>(
            A0, K0, A1, K1, pw, b, gg, bb, rin, rout, o, n, doLN);
    };

    // ---- edge degree counts (same for both layers) ----
    hipMemsetAsync(cntc, 0, (size_t)C_N * 4, stream);
    hipMemsetAsync(cntv, 0, (size_t)V_N * 4, stream);
    count_kernel<<<dim3((E_N + 255) / 256), blk, 0, stream>>>(ci, cntc, E_N);
    count_kernel<<<dim3((E_N + 255) / 256), blk, 0, stream>>>(vi, cntv, E_N);

    // ---- embeddings ----
    emb1_kernel<21><<<dim3((V_N + 7) / 8), blk, 0, stream>>>(var_feats, V_N, vw1, vb1, vg1, vbt1, lin);
    gemm(lin, 256, nullptr, 0, vw2, pw_lin, vb2, vg2, vbt2, nullptr, nullptr, vh, V_N, 1);
    emb1_kernel<6><<<dim3((C_N + 7) / 8), blk, 0, stream>>>(con_feats, C_N, cw1, cb1, cg1, cbt1, lin);
    gemm(lin, 256, nullptr, 0, cw2, pw_lin, cb2, cg2, cbt2, nullptr, nullptr, ch, C_N, 1);

    // ---- GCN layers ----
    float* vout_final = (float*)d_out;                     // [V*256]
    float* cout_final = vout_final + (size_t)V_N * H_N;    // [C*256]
    float* vcur = vh; float* ccur = ch;
    for (int l = 0; l < 2; l++) {
        const Layer& P = L[l];
        float* vnext = (l == 0) ? vh2 : vout_final;
        float* cnext = (l == 0) ? ch2 : cout_final;

        // var -> con
        gemm(vcur, 256, nullptr, 0, P.lw_vc, pw_lin, P.lb_vc,
             nullptr, nullptr, nullptr, nullptr, lin, V_N, 0);
        hipMemsetAsync(aggc, 0, (size_t)C_N * H_N * 4, stream);
        scatter_kernel<<<dim3((E_N + 7) / 8), blk, 0, stream>>>(
            lin, vi, ci, edge_val, P.gw_vc, P.gb_vc, aggc, E_N);
        norm_kernel<<<dim3(C_N), blk, 0, stream>>>(aggc, cntc, C_N);
        // ch_new = relu(LN([aggc, ch_old] @ uw_vc + ub));  ch_next = ch_old + ch_new
        gemm(aggc, 256, ccur, 256, P.uw_vc, pw_up, P.ub_vc,
             P.ug_vc, P.ut_vc, ccur, cnext, chn, C_N, 1);

        // con -> var (uses ch_new, old vh)
        gemm(chn, 256, nullptr, 0, P.lw_cv, pw_lin, P.lb_cv,
             nullptr, nullptr, nullptr, nullptr, lin, C_N, 0);
        hipMemsetAsync(aggv, 0, (size_t)V_N * H_N * 4, stream);
        scatter_kernel<<<dim3((E_N + 7) / 8), blk, 0, stream>>>(
            lin, ci, vi, edge_val, P.gw_cv, P.gb_cv, aggv, E_N);
        norm_kernel<<<dim3(V_N), blk, 0, stream>>>(aggv, cntv, V_N);
        // vh_new (scratch=lin) ; vh_next = vh_old + vh_new
        gemm(aggv, 256, vcur, 256, P.uw_cv, pw_up, P.ub_cv,
             P.ug_cv, P.ut_cv, vcur, vnext, lin, V_N, 1);

        vcur = vnext; ccur = cnext;
    }
}
